// Backbone_30090540875922
// MI455X (gfx1250) — matrix operand order repaired and checked by reading, not compiled, not run on hardware
//
#include <hip/hip_runtime.h>
#include <hip/hip_bf16.h>

// ---------------------------------------------------------------------------
// GATv2 backbone for MI455X (gfx1250, wave32).
// GEMMs via v_wmma_f32_16x16x32_f16, A-tile staged with async global->LDS
// (ASYNCcnt path); edge phase via fast fp32 global atomics.
// ---------------------------------------------------------------------------

typedef __attribute__((ext_vector_type(16))) _Float16 v16h;
typedef __attribute__((ext_vector_type(8)))  float    v8f;

#define GN    50000
#define GE    1000000
#define GH    4
#define GC    32
#define GHC   128
#define GFIN  64
#define GFOUT 64

// ---------------------------------------------------------------------------
// WMMA GEMM:  Y[16*grid.x rows, NCOLS] = X[rows, KDIM] @ W[KDIM, NCOLS] (+bias)
// One wave per 16x16 output tile; block covers one 16-row strip x all NCOLS.
// A tile (f32) staged via global_load_async_to_lds_b128; W transposed+converted
// to f16 in LDS; K-loop = KDIM/32 WMMA ops per wave.
// ---------------------------------------------------------------------------
template<int KDIM, int NCOLS>
__global__ void gemm_wmma(const float* __restrict__ X,
                          const float* __restrict__ W,
                          const float* __restrict__ bias,
                          float* __restrict__ Y) {
    __shared__ _Float16 Wt[NCOLS][KDIM];   // Wt[n][k] = W[k*NCOLS + n]
    __shared__ float    Af[16][KDIM];      // Af[r][k] = X[row0+r][k]  (async-staged)

    const int tid  = threadIdx.x;
    const int nthr = NCOLS * 2;            // blockDim.x
    const long row0 = (long)blockIdx.x * 16;

    // ---- async stage of the 16 x KDIM f32 A tile into LDS (ASYNCcnt) ----
    for (int i = tid; i < 16 * (KDIM / 4); i += nthr) {
        const int r = i / (KDIM / 4);
        const int c = (i % (KDIM / 4)) * 4;
        const float* g = X + (row0 + r) * KDIM + c;
        // generic LDS pointer -> low 32 bits are the LDS byte address
        unsigned lds = (unsigned)(unsigned long long)(&Af[r][c]);
        asm volatile("global_load_async_to_lds_b128 %0, %1, off"
                     :: "v"(lds), "v"(g) : "memory");
    }

    // ---- transpose + f32->f16 convert W into LDS (needs VALU anyway) ----
    for (int i = tid; i < KDIM * NCOLS; i += nthr) {
        int k = i / NCOLS, n = i % NCOLS;
        Wt[n][k] = (_Float16)W[i];
    }

    asm volatile("s_wait_asynccnt 0x0" ::: "memory");
    __syncthreads();

    const int lane = tid & 31;
    const int wave = tid >> 5;
    const int n0   = wave * 16;
    const int rr   = lane & 15;   // row (A) / col (B) within tile
    const int kh   = lane >> 4;   // lane-half selects K sub-range

    v8f acc = {};
    #pragma unroll
    for (int k0 = 0; k0 < KDIM; k0 += 32) {
        v16h a, b;
        #pragma unroll
        for (int v = 0; v < 8; ++v) {
            // ISA 16-bit A 16x32 layout: V0-3 -> K = 8*kh + 2v, V4-7 -> K = 16 + 8*kh + 2(v-4)
            int ka = k0 + ((v < 4) ? (8 * kh + 2 * v) : (16 + 8 * kh + 2 * (v - 4)));
            a[2 * v]     = (_Float16)Af[rr][ka];
            a[2 * v + 1] = (_Float16)Af[rr][ka + 1];
            // B 32x16 layout (same K order as A): V0-3 -> K = 8*kh + 2v, V4-7 -> K = 16 + 8*kh + 2(v-4), N = lane&15
            int kb = ka;
            b[2 * v]     = Wt[n0 + rr][kb];
            b[2 * v + 1] = Wt[n0 + rr][kb + 1];
        }
        acc = __builtin_amdgcn_wmma_f32_16x16x32_f16(
                  false, a, false, b, (short)0, acc, false, false);
    }

    // ---- epilogue: hoisted bias (ocol is invariant across r) ----
    const int ocol = n0 + rr;
    const float bv = bias ? bias[ocol] : 0.0f;
    float* ybase = Y + row0 * NCOLS + ocol;
    #pragma unroll
    for (int r = 0; r < 8; ++r) {
        int orow = r + 8 * kh;   // D layout: VGPR r -> M=r (lanes 0-15) / r+8 (16-31)
        ybase[(long)orow * NCOLS] = acc[r] + bv;
    }
}

// ---------------------------------------------------------------------------
// Edge phase
// ---------------------------------------------------------------------------

// per-layer state init: mKeys=-inf key, s=0, agg=0
__global__ void init_layer(int* __restrict__ mKeys, float* __restrict__ ssum,
                           float* __restrict__ agg, long NH, long Nagg) {
    long i = (long)blockIdx.x * blockDim.x + threadIdx.x;
    if (i < NH) { mKeys[i] = (int)0x80000000; ssum[i] = 0.0f; }
    if (i < Nagg) agg[i] = 0.0f;
}

__device__ __forceinline__ int float_to_ordered(float f) {
    int b = __float_as_int(f);
    return (b >= 0) ? b : (b ^ 0x7fffffff);
}

// one thread per (edge, head): logit = att . leakyrelu(xl[src]+xr[dst]); atomic max per dst
__global__ void edge_logits(const int* __restrict__ src, const int* __restrict__ dst,
                            const float* __restrict__ xl, const float* __restrict__ xr,
                            const float* __restrict__ att,
                            float* __restrict__ logit, int* __restrict__ mKeys, long EH) {
    long t = (long)blockIdx.x * blockDim.x + threadIdx.x;
    if (t >= EH) return;
    const int  h = (int)(t & (GH - 1));
    const long e = t >> 2;
    const int  s = src[e], d = dst[e];
    const float* pl = xl + (long)s * GHC + h * GC;
    const float* pr = xr + (long)d * GHC + h * GC;
    const float* pa = att + h * GC;
    float acc = 0.0f;
    #pragma unroll
    for (int c = 0; c < GC; ++c) {
        float v = pl[c] + pr[c];
        v = (v > 0.0f) ? v : 0.2f * v;
        acc = fmaf(pa[c], v, acc);
    }
    logit[t] = acc;
    atomicMax(&mKeys[(long)d * GH + h], float_to_ordered(acc));
}

// decode ordered-int max keys -> float m (0 for empty / non-finite segments)
__global__ void decode_max(const int* __restrict__ mKeys, float* __restrict__ m, long NH) {
    long i = (long)blockIdx.x * blockDim.x + threadIdx.x;
    if (i >= NH) return;
    int k = mKeys[i];
    float v = 0.0f;
    if (k != (int)0x80000000) {
        int b = (k >= 0) ? k : (k ^ 0x7fffffff);
        v = __int_as_float(b);
        if (!isfinite(v)) v = 0.0f;
    }
    m[i] = v;
}

// per (edge, head): a = exp(logit - m[dst]); segment sum via fast fp atomic
__global__ void edge_exp(const int* __restrict__ dst, const float* __restrict__ m,
                         float* __restrict__ logit, float* __restrict__ ssum, long EH) {
    long t = (long)blockIdx.x * blockDim.x + threadIdx.x;
    if (t >= EH) return;
    const int  h = (int)(t & (GH - 1));
    const long e = t >> 2;
    const int  d = dst[e];
    float a = __expf(logit[t] - m[(long)d * GH + h]);
    logit[t] = a;                               // reuse buffer for a
    unsafeAtomicAdd(&ssum[(long)d * GH + h], a);
}

// one wave per edge: lane handles 4 contiguous channels (one float4 of xl[src])
__global__ void edge_aggregate(const int* __restrict__ src, const int* __restrict__ dst,
                               const float* __restrict__ xl, const float* __restrict__ a,
                               const float* __restrict__ ssum, float* __restrict__ agg,
                               long E) {
    const int lane = threadIdx.x & 31;
    const long e = (long)blockIdx.x * (blockDim.x >> 5) + (threadIdx.x >> 5);
    if (e >= E) return;
    const int s = src[e], d = dst[e];
    const int hc0 = lane * 4;          // channels hc0..hc0+3, all in head h
    const int h   = lane >> 3;
    const float alpha = a[e * GH + h] / (ssum[(long)d * GH + h] + 1e-16f);
    const float4 x = *(const float4*)(xl + (long)s * GHC + hc0);
    float* out = agg + (long)d * GHC + hc0;
    unsafeAtomicAdd(out + 0, alpha * x.x);
    unsafeAtomicAdd(out + 1, alpha * x.y);
    unsafeAtomicAdd(out + 2, alpha * x.z);
    unsafeAtomicAdd(out + 3, alpha * x.w);
}

// f = relu(agg + bias)
__global__ void finalize_relu(const float* __restrict__ agg, const float* __restrict__ bias,
                              float* __restrict__ f, long n) {
    long i = (long)blockIdx.x * blockDim.x + threadIdx.x;
    if (i >= n) return;
    float v = agg[i] + bias[i & (GHC - 1)];
    f[i] = (v > 0.0f) ? v : 0.0f;
}

// f += 0.5 * relu(agg + bias)
__global__ void finalize_residual(const float* __restrict__ agg, const float* __restrict__ bias,
                                  float* __restrict__ f, long n) {
    long i = (long)blockIdx.x * blockDim.x + threadIdx.x;
    if (i >= n) return;
    float v = agg[i] + bias[i & (GHC - 1)];
    v = (v > 0.0f) ? v : 0.0f;
    f[i] = f[i] + 0.5f * v;
}

// ---------------------------------------------------------------------------
// Host orchestration (all on `stream`, graph-capture safe)
// ---------------------------------------------------------------------------
extern "C" void kernel_launch(void* const* d_in, const int* in_sizes, int n_in,
                              void* d_out, int out_size, void* d_ws, size_t ws_size,
                              hipStream_t stream) {
    const float* data = (const float*)d_in[0];
    const int*   edges = (const int*)d_in[1];
    const float* Wl1 = (const float*)d_in[2];
    const float* bl1 = (const float*)d_in[3];
    const float* Wr1 = (const float*)d_in[4];
    const float* br1 = (const float*)d_in[5];
    const float* att1 = (const float*)d_in[6];
    const float* b1  = (const float*)d_in[7];
    const float* Wl2 = (const float*)d_in[8];
    const float* bl2 = (const float*)d_in[9];
    const float* Wr2 = (const float*)d_in[10];
    const float* br2 = (const float*)d_in[11];
    const float* att2 = (const float*)d_in[12];
    const float* b2  = (const float*)d_in[13];
    const float* Wd  = (const float*)d_in[14];
    float* out = (float*)d_out;

    const int* src = edges;
    const int* dst = edges + GE;

    // workspace carve-up (all 256B aligned by construction)
    char* w = (char*)d_ws;
    float* xl  = (float*)w; w += (size_t)GN * GHC * sizeof(float);
    float* xr  = (float*)w; w += (size_t)GN * GHC * sizeof(float);
    float* f   = (float*)w; w += (size_t)GN * GHC * sizeof(float);
    float* agg = (float*)w; w += (size_t)GN * GHC * sizeof(float);
    float* aE  = (float*)w; w += (size_t)GE * GH  * sizeof(float);  // logits, then a
    int*   mK  = (int*)  w; w += (size_t)GN * GH  * sizeof(int);
    float* mV  = (float*)w; w += (size_t)GN * GH  * sizeof(float);
    float* sS  = (float*)w; w += (size_t)GN * GH  * sizeof(float);

    const long NH   = (long)GN * GH;        // 200k
    const long Nagg = (long)GN * GHC;       // 6.4M
    const long EH   = (long)GE * GH;        // 4M
    const int  gemmBlocks = GN / 16;        // 3125 (N is a multiple of 16)
    const dim3 b256(256);
    const dim3 gInit((unsigned)((Nagg + 255) / 256));
    const dim3 gEH((unsigned)((EH + 255) / 256));
    const dim3 gNH((unsigned)((NH + 255) / 256));
    const dim3 gAggE((unsigned)(GE / 8));   // 8 waves (edges) per 256-thread block
    const dim3 gElem((unsigned)((Nagg + 255) / 256));

    // ---- layer 1 (K = F_IN = 64) ----
    gemm_wmma<GFIN, GHC><<<gemmBlocks, 256, 0, stream>>>(data, Wl1, bl1, xl);
    gemm_wmma<GFIN, GHC><<<gemmBlocks, 256, 0, stream>>>(data, Wr1, br1, xr);
    init_layer<<<gInit, b256, 0, stream>>>(mK, sS, agg, NH, Nagg);
    edge_logits<<<gEH, b256, 0, stream>>>(src, dst, xl, xr, att1, aE, mK, EH);
    decode_max<<<gNH, b256, 0, stream>>>(mK, mV, NH);
    edge_exp<<<gEH, b256, 0, stream>>>(dst, mV, aE, sS, EH);
    edge_aggregate<<<gAggE, b256, 0, stream>>>(src, dst, xl, aE, sS, agg, GE);
    finalize_relu<<<gElem, b256, 0, stream>>>(agg, b1, f, Nagg);

    // ---- 3 residual blocks (shared layer-2 weights, K = 128) ----
    for (int t = 0; t < 3; ++t) {
        gemm_wmma<GHC, GHC><<<gemmBlocks, 256, 0, stream>>>(f, Wl2, bl2, xl);
        gemm_wmma<GHC, GHC><<<gemmBlocks, 256, 0, stream>>>(f, Wr2, br2, xr);
        init_layer<<<gInit, b256, 0, stream>>>(mK, sS, agg, NH, Nagg);
        edge_logits<<<gEH, b256, 0, stream>>>(src, dst, xl, xr, att2, aE, mK, EH);
        decode_max<<<gNH, b256, 0, stream>>>(mK, mV, NH);
        edge_exp<<<gEH, b256, 0, stream>>>(dst, mV, aE, sS, EH);
        edge_aggregate<<<gAggE, b256, 0, stream>>>(src, dst, xl, aE, sS, agg, GE);
        finalize_residual<<<gElem, b256, 0, stream>>>(agg, b2, f, Nagg);
    }

    // ---- final dense 128 -> 64, no bias ----
    gemm_wmma<GHC, GFOUT><<<gemmBlocks, 128, 0, stream>>>(f, Wd, nullptr, out);
}
